// Transformer_86131274154405
// MI455X (gfx1250) — matrix operand order repaired and checked by reading, not compiled, not run on hardware
//
#include <hip/hip_runtime.h>
#include <math.h>

// ---------------------------------------------------------------------------
// CDNA5 (gfx1250) implementation of the 2-layer local/global ViT block.
// All dense GEMMs use v_wmma_f32_16x16x32_f16 (f16 in, f32 accumulate) with
// LDS-staged, double-buffered weight tiles shared by all 8 waves of a block.
// LN / softmax / GELU epilogues in fp32. Attention (49 or 64 keys) in VALU.
// ---------------------------------------------------------------------------

typedef _Float16 v16h __attribute__((ext_vector_type(16)));
typedef _Float16 v8h  __attribute__((ext_vector_type(8)));
typedef float    v8f  __attribute__((ext_vector_type(8)));

#define DEPTH 2
#define DIMC  256
#define NHEAD 8
#define DHEAD 64
#define INNER 512
#define HID   1024
#define PW    7
#define KGS   7
#define BATCH 8
#define SDIM  56
#define MROWS (BATCH * SDIM * SDIM)   // 25088
#define GDIM  (SDIM / KGS)            // 8
#define KGLOB (KGS * KGS * DIMC)      // 12544
#define BK    32                      // K chunk per WMMA

enum { EPI_BIAS = 1, EPI_GELU = 2, EPI_RESID = 4 };

union frag { v16h v; v8h h[2]; };

// ---------------------------------------------------------------------------
// GEMM: C[M,N] = A[M,K] (f16, row-major) x Bt[N,K] (f16, row-major = B^T)
// Block: 8 waves -> 128(M) x 64(N) tile. Each wave: 16(M) x 64(N) strip via
// 4 accumulators sharing one A fragment. B chunk (64x32 f16 = 4KB) is staged
// in LDS (double-buffered), loaded once per block and read by all 8 waves.
// All 4 B fragments are pulled into distinct registers before the WMMA group
// so the 8 ds_load_b128s stay in flight together (no per-WMMA dscnt stall).
// Requires M%128==0, N%64==0, K%32==0 (true for every call site here).
// ---------------------------------------------------------------------------
__global__ __launch_bounds__(256)
void gemm_wmma_f16(const _Float16* __restrict__ A, const _Float16* __restrict__ Bt,
                   const float* __restrict__ bias, _Float16* __restrict__ out16,
                   float* __restrict__ resid, int M, int N, int K, int flags)
{
    __shared__ _Float16 bsh[2][64 * BK];          // 2 x 4KB double buffer

    const int t     = threadIdx.x;
    const int lane  = t & 31;
    const int wid   = t >> 5;                     // 0..7 -> M sub-tile
    const int tileM = blockIdx.y * 128 + wid * 16;
    const int tileN = blockIdx.x * 64;
    const int l15   = lane & 15;
    const int khalf = lane >> 4;                  // 0 or 1

    // A frag (16x32): lane row = l15; elems 0..7 -> K=khalf*16+0..7,
    //                              elems 8..15 -> K = khalf*16 + 8 + 0..7.
    const _Float16* arow = A + (long)(tileM + l15) * K;

    // Cooperative B staging: thread t loads 8 halfs of row (t>>2) at (t&3)*8.
    const int bn = t >> 2;                        // 0..63 (column of C)
    const int bk = (t & 3) * 8;
    const _Float16* bptr = Bt + (long)(tileN + bn) * K + bk;

    // Per-wave LDS read base: fragment nn at (nn*16 + l15)*BK + khalf*16.
    const int brd = l15 * BK + khalf * 16;

    v8f acc0 = (v8f){0,0,0,0,0,0,0,0};
    v8f acc1 = acc0, acc2 = acc0, acc3 = acc0;

    const int nIter = K / BK;

    // Preload chunk 0 into buffer 0.
    v8h breg = *(const v8h*)bptr;
    *(v8h*)&bsh[0][bn * BK + bk] = breg;
    __syncthreads();

    for (int it = 0; it < nIter; ++it) {
        const int cur = it & 1;
        const int kk  = it * BK;

        // Prefetch next B chunk into registers (global), overlap with compute.
        if (it + 1 < nIter)
            breg = *(const v8h*)(bptr + (it + 1) * BK);
        // Software prefetch of the A stream two chunks ahead (global_prefetch).
        if (it + 2 < nIter)
            __builtin_prefetch(arow + kk + 2 * BK, 0, 1);

        // Issue A global loads first so loadcnt overlaps the LDS reads.
        frag af;
        af.h[0] = *(const v8h*)(arow + kk + khalf * 16);
        af.h[1] = *(const v8h*)(arow + kk + khalf * 16 + 8);

        // Load all four B fragments into distinct registers, then compute.
        const _Float16* bb = &bsh[cur][0];
        frag bf0, bf1, bf2, bf3;
        bf0.h[0] = *(const v8h*)(bb + brd);
        bf0.h[1] = *(const v8h*)(bb + brd + 8);
        bf1.h[0] = *(const v8h*)(bb + 16 * BK + brd);
        bf1.h[1] = *(const v8h*)(bb + 16 * BK + brd + 8);
        bf2.h[0] = *(const v8h*)(bb + 32 * BK + brd);
        bf2.h[1] = *(const v8h*)(bb + 32 * BK + brd + 8);
        bf3.h[0] = *(const v8h*)(bb + 48 * BK + brd);
        bf3.h[1] = *(const v8h*)(bb + 48 * BK + brd + 8);

        acc0 = __builtin_amdgcn_wmma_f32_16x16x32_f16(
                   false, af.v, false, bf0.v, (short)0, acc0, false, false);
        acc1 = __builtin_amdgcn_wmma_f32_16x16x32_f16(
                   false, af.v, false, bf1.v, (short)0, acc1, false, false);
        acc2 = __builtin_amdgcn_wmma_f32_16x16x32_f16(
                   false, af.v, false, bf2.v, (short)0, acc2, false, false);
        acc3 = __builtin_amdgcn_wmma_f32_16x16x32_f16(
                   false, af.v, false, bf3.v, (short)0, acc3, false, false);

        // Stage next chunk into the other buffer; its previous contents were
        // consumed at iteration it-1, protected by that iteration's barrier.
        if (it + 1 < nIter)
            *(v8h*)&bsh[cur ^ 1][bn * BK + bk] = breg;
        __syncthreads();
    }

    // C/D layout: VGPR v -> row = tileM + v + khalf*8, col within frag = l15.
    const int mbase = tileM + khalf * 8;
    v8f accs[4] = { acc0, acc1, acc2, acc3 };
#pragma unroll
    for (int nn = 0; nn < 4; ++nn) {
        const int n = tileN + nn * 16 + l15;
        const float bnv = (flags & EPI_BIAS) ? bias[n] : 0.f;
#pragma unroll
        for (int v = 0; v < 8; ++v) {
            const int m = mbase + v;
            float val = accs[nn][v] + bnv;
            if (flags & EPI_RESID) {
                resid[(long)m * N + n] += val;
            } else {
                if (flags & EPI_GELU)
                    val = 0.5f * val * (1.0f + erff(val * 0.70710678118654752f));
                out16[(long)m * N + n] = (_Float16)val;
            }
        }
    }
}

// ---------------------------------------------------------------------------
// LayerNorm over last dim (256), fp32 in -> f16 out. One block per row.
// ---------------------------------------------------------------------------
__global__ __launch_bounds__(256)
void ln_kernel(const float* __restrict__ X, const float* __restrict__ g,
               const float* __restrict__ b, _Float16* __restrict__ out)
{
    __shared__ float s1[256], s2[256];
    const int row = blockIdx.x, t = threadIdx.x;
    const float x = X[(long)row * DIMC + t];
    s1[t] = x; s2[t] = x * x;
    __syncthreads();
    for (int s = 128; s > 0; s >>= 1) {
        if (t < s) { s1[t] += s1[t + s]; s2[t] += s2[t + s]; }
        __syncthreads();
    }
    const float mean = s1[0] * (1.f / DIMC);
    const float var  = s2[0] * (1.f / DIMC) - mean * mean;
    const float r    = rsqrtf(var + 1e-5f);
    out[(long)row * DIMC + t] = (_Float16)((x - mean) * r * g[t] + b[t]);
}

// ---------------------------------------------------------------------------
// Weight convert+transpose: W[K,N] fp32 -> Wt[N,K] f16 (coalesced writes).
// ---------------------------------------------------------------------------
__global__ void convert_w(const float* __restrict__ W, _Float16* __restrict__ Wt,
                          int K, int N)
{
    const long idx = (long)blockIdx.x * blockDim.x + threadIdx.x;
    if (idx >= (long)K * N) return;
    const int n = (int)(idx / K), k = (int)(idx % K);
    Wt[idx] = (_Float16)W[(long)k * N + n];
}

// ---------------------------------------------------------------------------
// Gather for global KV "conv": Ag[g, (p*7+q)*256 + c] = LH[row(b,Xg,p,Yg,q), c]
// ---------------------------------------------------------------------------
__global__ void gather_global(const _Float16* __restrict__ LH, _Float16* __restrict__ Ag)
{
    const long idx = (long)blockIdx.x * blockDim.x + threadIdx.x;
    if (idx >= (long)(BATCH * GDIM * GDIM) * KGLOB) return;
    const int g = (int)(idx / KGLOB), k = (int)(idx % KGLOB);
    const int p = k / (KGS * DIMC), rem = k % (KGS * DIMC);
    const int q = rem / DIMC, c = rem % DIMC;
    const int b = g >> 6, Xg = (g >> 3) & 7, Yg = g & 7;
    const int srow = b * (SDIM * SDIM) + (Xg * KGS + p) * SDIM + (Yg * KGS + q);
    Ag[idx] = LH[(long)srow * DIMC + c];
}

// ---------------------------------------------------------------------------
// Local windowed attention: 512 windows x 8 heads, 49 queries x 49 keys x 64.
// One thread per (window, head, query); online-softmax, fp32 accumulators.
// Q: [25088,512] f16; KV: [25088,1024] f16 (K | V); O: [25088,512] f16.
// ---------------------------------------------------------------------------
__global__ __launch_bounds__(64)
void local_attn(const _Float16* __restrict__ Q, const _Float16* __restrict__ KV,
                _Float16* __restrict__ O)
{
    const int wh = blockIdx.x;              // 512*8
    const int h = wh & 7, w = wh >> 3;
    const int b = w >> 6, Xw = (w >> 3) & 7, Yw = w & 7;
    const int p = threadIdx.x;
    if (p >= PW * PW) return;
    const int r = b * (SDIM * SDIM) + (Xw * PW + p / PW) * SDIM + (Yw * PW + p % PW);

    float q[DHEAD];
#pragma unroll
    for (int c = 0; c < DHEAD; ++c) q[c] = (float)Q[(long)r * INNER + h * DHEAD + c];

    float m = -1e30f, l = 0.f, acc[DHEAD];
#pragma unroll
    for (int c = 0; c < DHEAD; ++c) acc[c] = 0.f;

    for (int j = 0; j < PW * PW; ++j) {
        const int rj = b * (SDIM * SDIM) + (Xw * PW + j / PW) * SDIM + (Yw * PW + j % PW);
        const _Float16* kp = KV + (long)rj * (2 * INNER) + h * DHEAD;
        float d = 0.f;
#pragma unroll
        for (int c = 0; c < DHEAD; ++c) d += q[c] * (float)kp[c];
        d *= 0.125f;                              // DH^-0.5
        const float mn = fmaxf(m, d);
        const float ca = __expf(m - mn), cb = __expf(d - mn);
        l = l * ca + cb;
        const _Float16* vp = kp + INNER;
#pragma unroll
        for (int c = 0; c < DHEAD; ++c) acc[c] = acc[c] * ca + cb * (float)vp[c];
        m = mn;
    }
    const float inv = 1.f / l;
#pragma unroll
    for (int c = 0; c < DHEAD; ++c)
        O[(long)r * INNER + h * DHEAD + c] = (_Float16)(acc[c] * inv);
}

// ---------------------------------------------------------------------------
// Global attention: per (b,h) 3136 queries vs 64 keys. KVg: [512,1024] f16.
// ---------------------------------------------------------------------------
__global__ __launch_bounds__(256)
void global_attn(const _Float16* __restrict__ Q, const _Float16* __restrict__ KVg,
                 _Float16* __restrict__ O)
{
    const int idx = blockIdx.x * 256 + threadIdx.x;      // 8*8*3136 = 200704
    if (idx >= BATCH * NHEAD * SDIM * SDIM) return;
    const int b = idx / (NHEAD * SDIM * SDIM);
    const int rem = idx % (NHEAD * SDIM * SDIM);
    const int h = rem / (SDIM * SDIM), s = rem % (SDIM * SDIM);
    const int r = b * (SDIM * SDIM) + s;

    float q[DHEAD];
#pragma unroll
    for (int c = 0; c < DHEAD; ++c) q[c] = (float)Q[(long)r * INNER + h * DHEAD + c];

    float m = -1e30f, l = 0.f, acc[DHEAD];
#pragma unroll
    for (int c = 0; c < DHEAD; ++c) acc[c] = 0.f;

    for (int j = 0; j < GDIM * GDIM; ++j) {
        const _Float16* kp = KVg + (long)(b * GDIM * GDIM + j) * (2 * INNER) + h * DHEAD;
        float d = 0.f;
#pragma unroll
        for (int c = 0; c < DHEAD; ++c) d += q[c] * (float)kp[c];
        d *= 0.125f;
        const float mn = fmaxf(m, d);
        const float ca = __expf(m - mn), cb = __expf(d - mn);
        l = l * ca + cb;
        const _Float16* vp = kp + INNER;
#pragma unroll
        for (int c = 0; c < DHEAD; ++c) acc[c] = acc[c] * ca + cb * (float)vp[c];
        m = mn;
    }
    const float inv = 1.f / l;
#pragma unroll
    for (int c = 0; c < DHEAD; ++c)
        O[(long)r * INNER + h * DHEAD + c] = (_Float16)(acc[c] * inv);
}

// ---------------------------------------------------------------------------
// Host orchestration
// ---------------------------------------------------------------------------
static inline void run_gemm(const _Float16* A, const _Float16* Bt, const float* bias,
                            _Float16* out16, float* resid, int M, int N, int K,
                            int flags, hipStream_t s)
{
    dim3 grid(N / 64, M / 128);
    gemm_wmma_f16<<<grid, 256, 0, s>>>(A, Bt, bias, out16, resid, M, N, K, flags);
}

static inline void run_conv(const float* W, _Float16* Wt, int K, int N, hipStream_t s)
{
    long total = (long)K * N;
    convert_w<<<(unsigned)((total + 255) / 256), 256, 0, s>>>(W, Wt, K, N);
}

extern "C" void kernel_launch(void* const* d_in, const int* in_sizes, int n_in,
                              void* d_out, int out_size, void* d_ws, size_t ws_size,
                              hipStream_t stream)
{
    (void)in_sizes; (void)n_in; (void)out_size; (void)ws_size;

    const float* x_in  = (const float*)d_in[0];
    const float* ln_g  = (const float*)d_in[1];
    const float* ln_b  = (const float*)d_in[2];
    const float* la_wq = (const float*)d_in[3];
    const float* la_wkv= (const float*)d_in[4];
    const float* la_wo = (const float*)d_in[5];
    const float* la_bo = (const float*)d_in[6];
    const float* m1_w1 = (const float*)d_in[7];
    const float* m1_b1 = (const float*)d_in[8];
    const float* m1_w2 = (const float*)d_in[9];
    const float* m1_b2 = (const float*)d_in[10];
    const float* ga_wq = (const float*)d_in[11];
    const float* ga_wkv= (const float*)d_in[12];
    const float* ga_wo = (const float*)d_in[13];
    const float* ga_bo = (const float*)d_in[14];
    const float* m2_w1 = (const float*)d_in[15];
    const float* m2_b1 = (const float*)d_in[16];
    const float* m2_w2 = (const float*)d_in[17];
    const float* m2_b2 = (const float*)d_in[18];

    float* X = (float*)d_out;   // running activation, fp32 [25088,256]
    hipMemcpyAsync(X, x_in, (size_t)MROWS * DIMC * sizeof(float),
                   hipMemcpyDeviceToDevice, stream);

    // Workspace carve (f16 units): LH | A16 | Q16 | KV16 | W16  (~167 MB)
    _Float16* base = (_Float16*)d_ws;
    _Float16* LH   = base;                                       // 25088*256
    _Float16* A16  = LH  + (size_t)MROWS * DIMC;                 // 25088*1024
    _Float16* Q16  = A16 + (size_t)MROWS * HID;                  // 25088*512
    _Float16* KV16 = Q16 + (size_t)MROWS * INNER;                // 25088*1024
    _Float16* W16  = KV16 + (size_t)MROWS * HID;                 // 12544*1024

    for (int i = 0; i < DEPTH; ++i) {
        const float* g = ln_g + (size_t)i * 4 * DIMC;
        const float* b = ln_b + (size_t)i * 4 * DIMC;

        // ---- local attention block ----
        ln_kernel<<<MROWS, 256, 0, stream>>>(X, g + 0 * DIMC, b + 0 * DIMC, LH);
        run_conv(la_wq + (size_t)i * DIMC * INNER, W16, DIMC, INNER, stream);
        run_gemm(LH, W16, nullptr, Q16, nullptr, MROWS, INNER, DIMC, 0, stream);
        run_conv(la_wkv + (size_t)i * DIMC * 2 * INNER, W16, DIMC, 2 * INNER, stream);
        run_gemm(LH, W16, nullptr, KV16, nullptr, MROWS, 2 * INNER, DIMC, 0, stream);
        local_attn<<<512 * NHEAD, 64, 0, stream>>>(Q16, KV16, A16);
        run_conv(la_wo + (size_t)i * INNER * DIMC, W16, INNER, DIMC, stream);
        run_gemm(A16, W16, la_bo + (size_t)i * DIMC, nullptr, X,
                 MROWS, DIMC, INNER, EPI_BIAS | EPI_RESID, stream);

        // ---- MLP 1 ----
        ln_kernel<<<MROWS, 256, 0, stream>>>(X, g + 1 * DIMC, b + 1 * DIMC, LH);
        run_conv(m1_w1 + (size_t)i * DIMC * HID, W16, DIMC, HID, stream);
        run_gemm(LH, W16, m1_b1 + (size_t)i * HID, A16, nullptr,
                 MROWS, HID, DIMC, EPI_BIAS | EPI_GELU, stream);
        run_conv(m1_w2 + (size_t)i * HID * DIMC, W16, HID, DIMC, stream);
        run_gemm(A16, W16, m1_b2 + (size_t)i * DIMC, nullptr, X,
                 MROWS, DIMC, HID, EPI_BIAS | EPI_RESID, stream);

        // ---- global attention block ----
        ln_kernel<<<MROWS, 256, 0, stream>>>(X, g + 2 * DIMC, b + 2 * DIMC, LH);
        run_conv(ga_wq + (size_t)i * DIMC * INNER, W16, DIMC, INNER, stream);
        run_gemm(LH, W16, nullptr, Q16, nullptr, MROWS, INNER, DIMC, 0, stream);
        {
            long total = (long)(BATCH * GDIM * GDIM) * KGLOB;
            gather_global<<<(unsigned)((total + 255) / 256), 256, 0, stream>>>(LH, A16);
        }
        run_conv(ga_wkv + (size_t)i * KGLOB * 2 * INNER, W16, KGLOB, 2 * INNER, stream);
        run_gemm(A16, W16, nullptr, KV16, nullptr,
                 BATCH * GDIM * GDIM, 2 * INNER, KGLOB, 0, stream);
        global_attn<<<(BATCH * NHEAD * SDIM * SDIM) / 256, 256, 0, stream>>>(Q16, KV16, A16);
        run_conv(ga_wo + (size_t)i * INNER * DIMC, W16, INNER, DIMC, stream);
        run_gemm(A16, W16, ga_bo + (size_t)i * DIMC, nullptr, X,
                 MROWS, DIMC, INNER, EPI_BIAS | EPI_RESID, stream);

        // ---- MLP 2 ----
        ln_kernel<<<MROWS, 256, 0, stream>>>(X, g + 3 * DIMC, b + 3 * DIMC, LH);
        run_conv(m2_w1 + (size_t)i * DIMC * HID, W16, DIMC, HID, stream);
        run_gemm(LH, W16, m2_b1 + (size_t)i * HID, A16, nullptr,
                 MROWS, HID, DIMC, EPI_BIAS | EPI_GELU, stream);
        run_conv(m2_w2 + (size_t)i * HID * DIMC, W16, HID, DIMC, stream);
        run_gemm(A16, W16, m2_b2 + (size_t)i * DIMC, nullptr, X,
                 MROWS, DIMC, HID, EPI_BIAS | EPI_RESID, stream);
    }
}
